// MPNNMod_4355096838268
// MI455X (gfx1250) — compile-verified, matrix operand order checked by reading, not run on hardware
//
#include <hip/hip_runtime.h>
#include <hip/hip_bf16.h>

typedef __attribute__((ext_vector_type(16))) _Float16 v16h;
typedef __attribute__((ext_vector_type(8)))  float    v8f;

#define N_GRAPHS 128
#define NPG      128
#define N_NODES  16384
#define N_EDGES  131072
#define DM       256
#define H_HEADS  8
#define DH       32

#define LDA_T 40   // padded LDS row stride (f16 units); 40*2=80B keeps 16B chunks aligned

// ---------------------------------------------------------------------------
// fragment load helper: 16-bit A/B 16x16x32 layout (ISA 7.12.2).
// lane = {m|n} + 16*half ; elements 0..7 <- base[8*half .. +7], 8..15 <- base[16+8*half .. +7]
__device__ __forceinline__ v16h load_frag(const _Float16* base, int half) {
    v16h f;
    ((uint4*)&f)[0] = *(const uint4*)(base + 8 * half);
    ((uint4*)&f)[1] = *(const uint4*)(base + 16 + 8 * half);
    return f;
}

// stage one 32-wide B k-slab transposed into Bs[n][k]; thread -> k row tid>>3, 8 n
__device__ __forceinline__ void stage_B(_Float16* Bs, const _Float16* Wrow, int tid, int N) {
    int kr = tid >> 3;
    int nn = (tid & 7) * 8;
    _Float16 tmp[8];
    *(uint4*)tmp = *(const uint4*)(Wrow + (size_t)kr * N + nn);
#pragma unroll
    for (int i = 0; i < 8; ++i) Bs[(nn + i) * LDA_T + kr] = tmp[i];
}

// ---------------------------------------------------------------------------
// elementwise helpers
__global__ void cvt_f16_kernel(const float* __restrict__ x, _Float16* __restrict__ y, int n) {
    for (int i = blockIdx.x * blockDim.x + threadIdx.x; i < n; i += gridDim.x * blockDim.x)
        y[i] = (_Float16)x[i];
}

__global__ void addvn_kernel(const float* __restrict__ x, const float* __restrict__ vn,
                             _Float16* __restrict__ h, int n, int D) {
    for (int i = blockIdx.x * blockDim.x + threadIdx.x; i < n; i += gridDim.x * blockDim.x)
        h[i] = (_Float16)(x[i] + vn[i & (D - 1)]);
}

__global__ void add2_kernel(float* __restrict__ a, const float* __restrict__ b, int n) {
    for (int i = blockIdx.x * blockDim.x + threadIdx.x; i < n; i += gridDim.x * blockDim.x)
        a[i] += b[i];
}

// ---------------------------------------------------------------------------
// Generic WMMA GEMM: Y[M,N] = epi( A[M,K](f16) @ W[K,N](f16) + bias )
// block = 256 thr (8 waves, 4 row-groups x 2 col-groups), tile M=64 N=64, K-step 32.
// Each wave: 16x32 C (2 wmma / k-step, shared A fragment).
// Requires K % 32 == 0, M % 64 == 0, N % 64 == 0  -> branch-free staging.
// EPI: 0 -> f32 store ; 3 -> f16 store ; 4 -> relu + f16 store
template <int EPI>
__global__ void gemm_wmma(const _Float16* __restrict__ A, const _Float16* __restrict__ W,
                          const float* __restrict__ bias, void* __restrict__ Y,
                          int M, int N, int K) {
    __shared__ __align__(16) _Float16 As[64 * LDA_T];   // [m][k]
    __shared__ __align__(16) _Float16 Bs[64 * LDA_T];   // transposed: [n][k]
    const int tid  = threadIdx.x;
    const int lane = tid & 31;
    const int wave = tid >> 5;
    const int bm = blockIdx.x * 64;
    const int bn = blockIdx.y * 64;
    const int wm = (wave >> 1) * 16;
    const int wn = (wave & 1) * 32;
    const int mrow = lane & 15;
    const int half = lane >> 4;
    const int ar = tid >> 2;           // A-stage row
    const int ak = (tid & 3) * 8;      // A-stage k chunk
    const _Float16* Arow = A + (size_t)(bm + ar) * K + ak;
    v8f acc[2] = {};
    for (int k0 = 0; k0 < K; k0 += 32) {
        *(uint4*)&As[ar * LDA_T + ak] = *(const uint4*)(Arow + k0);
        if (k0 + 32 < K) __builtin_prefetch(Arow + k0 + 32, 0, 1);
        stage_B(Bs, W + (size_t)k0 * N + bn, tid, N);
        __syncthreads();
        v16h a = load_frag(&As[(wm + mrow) * LDA_T], half);
#pragma unroll
        for (int s = 0; s < 2; ++s) {
            v16h b = load_frag(&Bs[(wn + s * 16 + mrow) * LDA_T], half);
            acc[s] = __builtin_amdgcn_wmma_f32_16x16x32_f16(false, a, false, b, (short)0,
                                                            acc[s], false, false);
        }
        __syncthreads();
    }
#pragma unroll
    for (int s = 0; s < 2; ++s) {
#pragma unroll
        for (int r = 0; r < 8; ++r) {
            int gm = bm + wm + r + 8 * half;
            int gn = bn + wn + s * 16 + mrow;
            float v = acc[s][r] + bias[gn];
            if (EPI == 4) v = v > 0.f ? v : 0.f;
            if (EPI == 0) ((float*)Y)[(size_t)gm * N + gn] = v;
            else          ((_Float16*)Y)[(size_t)gm * N + gn] = (_Float16)v;
        }
    }
}

// ---------------------------------------------------------------------------
// Edge message GEMM with gathered A rows and fused relu + scatter-add epilogue:
//   msg[e,:] = relu([h[src[e]] || eattr[e]] @ W + b) ;  agg[dst[e],:] += msg[e,:]
// inDim % 32 == 0 (128 or 256), edge dim = 16 -> main k-loop is branch-free over
// node features; one fixed tail step handles the 16 edge-attr columns (zero-padded
// to a 32-wide slab).
__global__ void gemm_msg(const _Float16* __restrict__ Hm, int inDim,
                         const _Float16* __restrict__ Ea,
                         const int* __restrict__ srcI, const int* __restrict__ dstI,
                         const _Float16* __restrict__ W, const float* __restrict__ bias,
                         float* __restrict__ Agg, int N) {
    __shared__ __align__(16) _Float16 As[64 * LDA_T];
    __shared__ __align__(16) _Float16 Bs[64 * LDA_T];
    __shared__ int sSrc[64];
    __shared__ int sDst[64];
    const int tid  = threadIdx.x;
    const int lane = tid & 31;
    const int wave = tid >> 5;
    const int bm = blockIdx.x * 64;
    const int bn = blockIdx.y * 64;
    const int wm = (wave >> 1) * 16;
    const int wn = (wave & 1) * 32;
    const int mrow = lane & 15;
    const int half = lane >> 4;
    const int ar = tid >> 2;
    const int ak = (tid & 3) * 8;
    if (tid < 64) { sSrc[tid] = srcI[bm + tid]; sDst[tid] = dstI[bm + tid]; }
    __syncthreads();
    const _Float16* Hrow = Hm + (size_t)sSrc[ar] * inDim + ak;
    v8f acc[2] = {};
    // ---- main loop: node-feature columns, branch-free ----
    for (int k0 = 0; k0 < inDim; k0 += 32) {
        *(uint4*)&As[ar * LDA_T + ak] = *(const uint4*)(Hrow + k0);
        stage_B(Bs, W + (size_t)k0 * N + bn, tid, N);
        __syncthreads();
        v16h a = load_frag(&As[(wm + mrow) * LDA_T], half);
#pragma unroll
        for (int s = 0; s < 2; ++s) {
            v16h b = load_frag(&Bs[(wn + s * 16 + mrow) * LDA_T], half);
            acc[s] = __builtin_amdgcn_wmma_f32_16x16x32_f16(false, a, false, b, (short)0,
                                                            acc[s], false, false);
        }
        __syncthreads();
    }
    // ---- tail step: 16 edge-attr columns, zero-padded to 32 ----
    {
        uint4 av = {0u, 0u, 0u, 0u};
        if (ak < 16) av = *(const uint4*)(Ea + (size_t)(bm + ar) * 16 + ak);
        *(uint4*)&As[ar * LDA_T + ak] = av;
        int kr = tid >> 3;
        int nn = (tid & 7) * 8;
        _Float16 tmp[8];
        uint4 zv = {0u, 0u, 0u, 0u};
        *(uint4*)tmp = (kr < 16) ? *(const uint4*)(W + (size_t)(inDim + kr) * N + bn + nn) : zv;
#pragma unroll
        for (int i = 0; i < 8; ++i) Bs[(nn + i) * LDA_T + kr] = tmp[i];
        __syncthreads();
        v16h a = load_frag(&As[(wm + mrow) * LDA_T], half);
#pragma unroll
        for (int s = 0; s < 2; ++s) {
            v16h b = load_frag(&Bs[(wn + s * 16 + mrow) * LDA_T], half);
            acc[s] = __builtin_amdgcn_wmma_f32_16x16x32_f16(false, a, false, b, (short)0,
                                                            acc[s], false, false);
        }
    }
    // ---- relu + atomic scatter-add epilogue ----
#pragma unroll
    for (int s = 0; s < 2; ++s) {
#pragma unroll
        for (int r = 0; r < 8; ++r) {
            int rowInBlk = wm + r + 8 * half;
            int gn = bn + wn + s * 16 + mrow;
            float v = acc[s][r] + bias[gn];
            v = v > 0.f ? v : 0.f;
            __hip_atomic_fetch_add(&Agg[(size_t)sDst[rowInBlk] * N + gn], v,
                                   __ATOMIC_RELAXED, __HIP_MEMORY_SCOPE_AGENT);
        }
    }
}

// ---------------------------------------------------------------------------
// Row LayerNorm (+optional residual, relu); one wave per row.
__global__ void ln_kernel(const float* __restrict__ X, const float* __restrict__ R,
                          const float* __restrict__ g, const float* __restrict__ bt,
                          float* __restrict__ O32, int s32,
                          _Float16* __restrict__ O16, int s16,
                          int M, int D, int doRelu) {
    int row = blockIdx.x * 8 + (threadIdx.x >> 5);
    if (row >= M) return;
    int lane = threadIdx.x & 31;
    const float* x = X + (size_t)row * D;
    const float* r = R ? (R + (size_t)row * D) : nullptr;
    float s = 0.f, ss = 0.f;
    for (int c = lane; c < D; c += 32) {
        float v = x[c] + (r ? r[c] : 0.f);
        s += v; ss += v * v;
    }
    for (int o = 16; o > 0; o >>= 1) { s += __shfl_xor(s, o, 32); ss += __shfl_xor(ss, o, 32); }
    float mean = s / (float)D;
    float var  = ss / (float)D - mean * mean;
    float rstd = rsqrtf(var + 1e-5f);
    for (int c = lane; c < D; c += 32) {
        float v = x[c] + (r ? r[c] : 0.f);
        float y = (v - mean) * rstd * g[c] + bt[c];
        if (doRelu) y = y > 0.f ? y : 0.f;
        if (O32) O32[(size_t)row * s32 + c] = y;
        if (O16) O16[(size_t)row * s16 + c] = (_Float16)y;
    }
}

// ---------------------------------------------------------------------------
// Attention scores: S[b,h,q,k] = (Q[b,q,h,:].K[b,k,h,:]) * scale ; dh=32 = one WMMA.
// grid (qtile=8, b*h=1024), block 256 (wave -> ktile).
__global__ void attn_scores(const _Float16* __restrict__ Q, const _Float16* __restrict__ Km,
                            float* __restrict__ S, float scale) {
    int bh = blockIdx.y; int b = bh >> 3; int h = bh & 7;
    int q0 = blockIdx.x * 16;
    int k0 = (threadIdx.x >> 5) * 16;
    int lane = threadIdx.x & 31;
    int mrow = lane & 15, half = lane >> 4;
    const _Float16* qp = Q  + ((size_t)(b * NPG + q0 + mrow)) * DM + h * DH;
    const _Float16* kp = Km + ((size_t)(b * NPG + k0 + mrow)) * DM + h * DH;
    v16h a  = load_frag(qp, half);
    v16h bb = load_frag(kp, half);
    v8f acc = {};
    acc = __builtin_amdgcn_wmma_f32_16x16x32_f16(false, a, false, bb, (short)0, acc,
                                                 false, false);
#pragma unroll
    for (int r = 0; r < 8; ++r) {
        int q = q0 + r + 8 * half;
        int k = k0 + mrow;
        S[((size_t)bh * NPG + q) * NPG + k] = acc[r] * scale;
    }
}

// softmax over 128-wide rows; one wave per row; f16 output
__global__ void softmax_rows(const float* __restrict__ S, _Float16* __restrict__ P) {
    int row = blockIdx.x * 8 + (threadIdx.x >> 5);
    int lane = threadIdx.x & 31;
    const float* s = S + (size_t)row * NPG;
    float v[4], mx = -1e30f;
#pragma unroll
    for (int i = 0; i < 4; ++i) { v[i] = s[lane + 32 * i]; mx = fmaxf(mx, v[i]); }
    for (int o = 16; o > 0; o >>= 1) mx = fmaxf(mx, __shfl_xor(mx, o, 32));
    float sum = 0.f;
#pragma unroll
    for (int i = 0; i < 4; ++i) { v[i] = __expf(v[i] - mx); sum += v[i]; }
    for (int o = 16; o > 0; o >>= 1) sum += __shfl_xor(sum, o, 32);
    float inv = 1.f / sum;
#pragma unroll
    for (int i = 0; i < 4; ++i)
        P[(size_t)row * NPG + lane + 32 * i] = (_Float16)(v[i] * inv);
}

// ctx[b,q,h,d] = sum_k P[b,h,q,k] V[b,k,h,d] ; grid (qtile=8, b*h), block 64 (wave->dtile)
__global__ void attn_av(const _Float16* __restrict__ P, const _Float16* __restrict__ V,
                        _Float16* __restrict__ C) {
    int bh = blockIdx.y; int b = bh >> 3; int h = bh & 7;
    int q0 = blockIdx.x * 16;
    int n0 = (threadIdx.x >> 5) * 16;
    int lane = threadIdx.x & 31;
    int mrow = lane & 15, half = lane >> 4;
    const _Float16* pr = P + ((size_t)bh * NPG + q0 + mrow) * NPG;
    v8f acc = {};
    for (int kk = 0; kk < NPG; kk += 32) {
        v16h a = load_frag(pr + kk, half);
        v16h bb;
#pragma unroll
        for (int v = 0; v < 8; ++v) {
            int kb = ((v < 4) ? 2 * v : 16 + 2 * (v - 4)) + 8 * half;
            bb[2 * v]     = V[((size_t)(b * NPG + kk + kb))     * DM + h * DH + n0 + mrow];
            bb[2 * v + 1] = V[((size_t)(b * NPG + kk + kb + 1)) * DM + h * DH + n0 + mrow];
        }
        acc = __builtin_amdgcn_wmma_f32_16x16x32_f16(false, a, false, bb, (short)0, acc,
                                                     false, false);
    }
#pragma unroll
    for (int r = 0; r < 8; ++r) {
        int q = q0 + r + 8 * half;
        int d = n0 + mrow;
        C[((size_t)(b * NPG + q)) * DM + h * DH + d] = (_Float16)acc[r];
    }
}

// mean-pool over NPG nodes per graph: block = one graph, thread = channel
__global__ void pool_mean(const float* __restrict__ X, float* __restrict__ out,
                          int ostride, int ooff) {
    int b = blockIdx.x, d = threadIdx.x;
    float s = 0.f;
    for (int l = 0; l < NPG; ++l) s += X[((size_t)b * NPG + l) * DM + d];
    out[(size_t)b * ostride + ooff + d] = s * (1.f / (float)NPG);
}

// y[row] = dot(H[row,:], w) + b0 ; one wave per row
__global__ void out_head(const float* __restrict__ Hx, const float* __restrict__ w,
                         const float* __restrict__ b0, float* __restrict__ y, int M, int D) {
    int row = blockIdx.x * 8 + (threadIdx.x >> 5);
    if (row >= M) return;
    int lane = threadIdx.x & 31;
    float s = 0.f;
    for (int c = lane; c < D; c += 32) s += Hx[(size_t)row * D + c] * w[c];
    for (int o = 16; o > 0; o >>= 1) s += __shfl_xor(s, o, 32);
    if (lane == 0) y[row] = s + b0[0];
}

// ---------------------------------------------------------------------------
extern "C" void kernel_launch(void* const* d_in, const int* in_sizes, int n_in,
                              void* d_out, int out_size, void* d_ws, size_t ws_size,
                              hipStream_t stream) {
    (void)in_sizes; (void)n_in; (void)out_size; (void)ws_size;
    const float* xin[2]   = {(const float*)d_in[0], (const float*)d_in[4]};
    const int*   ei[2]    = {(const int*)d_in[1],   (const int*)d_in[5]};
    const float* eattr[2] = {(const float*)d_in[2], (const float*)d_in[6]};
    // params pytree (sorted keys): ff_mol 8-11, ff_pred 12-15, final 16-23,
    // m2p 24-31, mol 32-72, norm_mol 73-74, norm_pred 75-76, out 77-78,
    // p2m 79-86, pred 87-127.
    auto P = [&](int i) { return (const float*)d_in[i]; };
    const int branchBase[2] = {32, 87};   // per-layer leaves: ln.b,ln.g,msg.b,msg.w,node.b,node.w,vn_emb,(+6 dead)
    const int mhaBase[2]    = {24, 79};   // bk,bo,bq,bv,wk,wo,wq,wv
    const int ffBase[2]     = {8, 12};    // l1.b,l1.w,l2.b,l2.w
    const int normB[2] = {73, 75}, normG[2] = {74, 76};

    char* wsB = (char*)d_ws;
    size_t off = 0;
    auto alloc = [&](size_t bytes) -> void* {
        void* r = wsB + off; off = (off + bytes + 255) & ~(size_t)255; return r;
    };
    auto cvt = [&](const float* src, _Float16* dst, size_t n) {
        int blocks = (int)((n + 2047) / 2048);
        cvt_f16_kernel<<<blocks, 256, 0, stream>>>(src, dst, (int)n);
    };
    auto cvtW = [&](int idx, size_t n) -> _Float16* {
        _Float16* w = (_Float16*)alloc(n * 2);
        cvt(P(idx), w, n);
        return w;
    };

    // ---- persistent: f16 weights ----
    _Float16 *msgw[2][3], *nodew[2][3], *projw[2];
    for (int b = 0; b < 2; ++b) {
        for (int l = 0; l < 3; ++l) {
            int in = l ? 256 : 128;
            msgw[b][l]  = cvtW(branchBase[b] + l * 13 + 3, (size_t)(in + 16) * 256);
            nodew[b][l] = cvtW(branchBase[b] + l * 13 + 5, (size_t)in * 256);
        }
        projw[b] = cvtW(branchBase[b] + 40, (size_t)768 * 256);
    }
    _Float16 *wk16[2], *wo16[2], *wq16[2], *wv16[2];
    for (int m = 0; m < 2; ++m) {
        wk16[m] = cvtW(mhaBase[m] + 4, 65536);
        wo16[m] = cvtW(mhaBase[m] + 5, 65536);
        wq16[m] = cvtW(mhaBase[m] + 6, 65536);
        wv16[m] = cvtW(mhaBase[m] + 7, 65536);
    }
    _Float16 *ffl1w[2], *ffl2w[2];
    for (int b = 0; b < 2; ++b) {
        ffl1w[b] = cvtW(ffBase[b] + 1, (size_t)256 * 1024);
        ffl2w[b] = cvtW(ffBase[b] + 3, (size_t)1024 * 256);
    }
    _Float16* fin0w = cvtW(17, (size_t)512 * 512);
    _Float16* fin1w = cvtW(21, (size_t)512 * 256);

    // ---- persistent activations ----
    float* nodef32[2]; _Float16* seqf16[2]; float* post[2];
    for (int b = 0; b < 2; ++b) {
        nodef32[b] = (float*)alloc((size_t)N_NODES * DM * 4);
        seqf16[b]  = (_Float16*)alloc((size_t)N_NODES * DM * 2);
        post[b]    = (float*)alloc((size_t)N_NODES * DM * 4);
    }
    float*    hcat   = (float*)alloc((size_t)N_GRAPHS * 512 * 4);
    _Float16* hcat16 = (_Float16*)alloc((size_t)N_GRAPHS * 512 * 2);
    float*    f0f32  = (float*)alloc((size_t)N_GRAPHS * 512 * 4);
    _Float16* f0f16  = (_Float16*)alloc((size_t)N_GRAPHS * 512 * 2);
    float*    f1f32  = (float*)alloc((size_t)N_GRAPHS * 256 * 4);
    float*    f1post = (float*)alloc((size_t)N_GRAPHS * 256 * 4);

    const size_t scratch0 = off;

    // =================== MPNN branches ===================
    for (int b = 0; b < 2; ++b) {
        size_t so = scratch0;
        auto sall = [&](size_t bytes) -> void* {
            void* r = wsB + so; so = (so + bytes + 255) & ~(size_t)255; return r;
        };
        _Float16* hf16 = (_Float16*)sall((size_t)N_NODES * 256 * 2);
        float*    xf32 = (float*)sall((size_t)N_NODES * 256 * 4);
        float*    agg  = (float*)sall((size_t)N_NODES * 256 * 4);
        float*    lin  = (float*)sall((size_t)N_NODES * 256 * 4);
        _Float16* jk   = (_Float16*)sall((size_t)N_NODES * 768 * 2);
        _Float16* ef16 = (_Float16*)sall((size_t)N_EDGES * 16 * 2);
        cvt(eattr[b], ef16, (size_t)N_EDGES * 16);
        const int* srcI = ei[b];
        const int* dstI = ei[b] + N_EDGES;
        int base = branchBase[b];
        for (int l = 0; l < 3; ++l) {
            int in = l ? 256 : 128;
            const float* xc = l ? xf32 : xin[b];
            size_t n = (size_t)N_NODES * in;
            addvn_kernel<<<(int)((n + 2047) / 2048), 256, 0, stream>>>(
                xc, P(base + l * 13 + 6), hf16, (int)n, in);
            hipMemsetAsync(agg, 0, (size_t)N_NODES * DM * 4, stream);
            gemm_msg<<<dim3(N_EDGES / 64, DM / 64), 256, 0, stream>>>(
                hf16, in, ef16, srcI, dstI, msgw[b][l], P(base + l * 13 + 2), agg, DM);
            gemm_wmma<0><<<dim3(N_NODES / 64, DM / 64), 256, 0, stream>>>(
                hf16, nodew[b][l], P(base + l * 13 + 4), lin, N_NODES, DM, in);
            ln_kernel<<<N_NODES / 8, 256, 0, stream>>>(
                lin, agg, P(base + l * 13 + 1), P(base + l * 13 + 0),
                xf32, 256, jk + l * 256, 768, N_NODES, DM, 1);
        }
        gemm_wmma<0><<<dim3(N_NODES / 64, DM / 64), 256, 0, stream>>>(
            jk, projw[b], P(base + 39), nodef32[b], N_NODES, DM, 768);
        cvt(nodef32[b], seqf16[b], (size_t)N_NODES * DM);
    }

    // =================== cross attention + FF (m2p then p2m) ===================
    for (int m = 0; m < 2; ++m) {           // m=0: q=mol kv=pred ; m=1: q=pred kv=mol
        int qb = m, kb = 1 - m;
        size_t so = scratch0;
        auto sall = [&](size_t bytes) -> void* {
            void* r = wsB + so; so = (so + bytes + 255) & ~(size_t)255; return r;
        };
        _Float16* qf = (_Float16*)sall((size_t)N_NODES * DM * 2);
        _Float16* kf = (_Float16*)sall((size_t)N_NODES * DM * 2);
        _Float16* vf = (_Float16*)sall((size_t)N_NODES * DM * 2);
        float*    sc = (float*)sall((size_t)N_GRAPHS * H_HEADS * NPG * NPG * 4);
        _Float16* pr = (_Float16*)sall((size_t)N_GRAPHS * H_HEADS * NPG * NPG * 2);
        _Float16* cx = (_Float16*)sall((size_t)N_NODES * DM * 2);
        float*    ao = (float*)sall((size_t)N_NODES * DM * 4);
        _Float16* fi = (_Float16*)sall((size_t)N_NODES * DM * 2);
        _Float16* f1 = (_Float16*)sall((size_t)N_NODES * 1024 * 2);
        float*    f2 = (float*)sall((size_t)N_NODES * DM * 4);
        int mb = mhaBase[m];
        gemm_wmma<3><<<dim3(N_NODES / 64, DM / 64), 256, 0, stream>>>(
            seqf16[qb], wq16[m], P(mb + 2), qf, N_NODES, DM, DM);
        gemm_wmma<3><<<dim3(N_NODES / 64, DM / 64), 256, 0, stream>>>(
            seqf16[kb], wk16[m], P(mb + 0), kf, N_NODES, DM, DM);
        gemm_wmma<3><<<dim3(N_NODES / 64, DM / 64), 256, 0, stream>>>(
            seqf16[kb], wv16[m], P(mb + 3), vf, N_NODES, DM, DM);
        attn_scores<<<dim3(NPG / 16, N_GRAPHS * H_HEADS), 256, 0, stream>>>(
            qf, kf, sc, 0.1767766952966369f /* 1/sqrt(32) */);
        softmax_rows<<<(N_GRAPHS * H_HEADS * NPG) / 8, 256, 0, stream>>>(sc, pr);
        attn_av<<<dim3(NPG / 16, N_GRAPHS * H_HEADS), 64, 0, stream>>>(pr, vf, cx);
        gemm_wmma<0><<<dim3(N_NODES / 64, DM / 64), 256, 0, stream>>>(
            cx, wo16[m], P(mb + 1), ao, N_NODES, DM, DM);
        ln_kernel<<<N_NODES / 8, 256, 0, stream>>>(
            nodef32[qb], ao, P(normG[qb]), P(normB[qb]),
            post[qb], 256, fi, 256, N_NODES, DM, 0);
        gemm_wmma<4><<<dim3(N_NODES / 64, 1024 / 64), 256, 0, stream>>>(
            fi, ffl1w[qb], P(ffBase[qb] + 0), f1, N_NODES, 1024, DM);
        gemm_wmma<0><<<dim3(N_NODES / 64, DM / 64), 256, 0, stream>>>(
            f1, ffl2w[qb], P(ffBase[qb] + 2), f2, N_NODES, DM, 1024);
        add2_kernel<<<(N_NODES * DM + 2047) / 2048, 256, 0, stream>>>(
            post[qb], f2, N_NODES * DM);
    }

    // =================== pooling + final MLP + head ===================
    pool_mean<<<N_GRAPHS, DM, 0, stream>>>(post[0], hcat, 512, 0);
    pool_mean<<<N_GRAPHS, DM, 0, stream>>>(post[1], hcat, 512, 256);
    cvt(hcat, hcat16, (size_t)N_GRAPHS * 512);
    gemm_wmma<0><<<dim3(N_GRAPHS / 64, 512 / 64), 256, 0, stream>>>(
        hcat16, fin0w, P(16), f0f32, N_GRAPHS, 512, 512);
    ln_kernel<<<N_GRAPHS / 8, 256, 0, stream>>>(
        f0f32, nullptr, P(19), P(18), nullptr, 0, f0f16, 512, N_GRAPHS, 512, 1);
    gemm_wmma<0><<<dim3(N_GRAPHS / 64, 256 / 64), 256, 0, stream>>>(
        f0f16, fin1w, P(20), f1f32, N_GRAPHS, 256, 512);
    ln_kernel<<<N_GRAPHS / 8, 256, 0, stream>>>(
        f1f32, nullptr, P(23), P(22), f1post, 256, nullptr, 0, N_GRAPHS, 256, 1);
    out_head<<<N_GRAPHS / 8, 256, 0, stream>>>(
        f1post, P(78), P(77), (float*)d_out, N_GRAPHS, 256);
}
